// SWinTr_12953621365011
// MI455X (gfx1250) — matrix operand order repaired and checked by reading, not compiled, not run on hardware
//
#include <hip/hip_runtime.h>
#include <hip/hip_bf16.h>
#include <math.h>

// ---------------------------------------------------------------------------
// Swin block on MI455X (gfx1250). All GEMMs use v_wmma_f32_16x16x32_bf16 with
// double-buffered global_load_async_to_lds_b128 staging (ASYNCcnt pipeline).
// N=16 H=W=64 C=512 NH=16 HD=32 WS=8 SHIFT=4 MLP_H=2048.
// ---------------------------------------------------------------------------

typedef __bf16 bf16;
typedef __attribute__((ext_vector_type(16))) __bf16 v16bf;
typedef __attribute__((ext_vector_type(8)))  __bf16 v8bf;
typedef __attribute__((ext_vector_type(8)))  float  v8f;

// LDS tile geometry: 128 rows/cols, 32 K per chunk, stride 40 elems (80 B)
// -> 16B-aligned v8bf reads, conflict-free bank pattern (20*lc mod 64 distinct).
#define TSTRIDE 40

__device__ __forceinline__ v8f wmma_bf16(v16bf a, v16bf b, v8f c) {
  return __builtin_amdgcn_wmma_f32_16x16x32_bf16(false, a, false, b, (short)0, c,
                                                 false, false);
}

// ---- global-memory fragment loaders (used by attention) -------------------
// A (16x32): lanes0-15 rows, VGPR0-3=K[kg*8..], VGPR4-7=K[16+kg*8..]
__device__ __forceinline__ v16bf frag_a(const bf16* __restrict__ base, int ld,
                                        int m0, int k0) {
  int lane = threadIdx.x & 31;
  const bf16* p = base + (long)(m0 + (lane & 15)) * ld + k0 + ((lane >> 4) << 3);
  v8bf lo = *(const v8bf*)p;
  v8bf hi = *(const v8bf*)(p + 16);
  v16bf r;
#pragma unroll
  for (int i = 0; i < 8; ++i) { r[i] = lo[i]; r[i + 8] = hi[i]; }
  return r;
}

// B (32x16) from row-major [N,K] weight: lane n holds K[kg*8..] and K[16+kg*8..] of row n.
__device__ __forceinline__ v16bf frag_b(const bf16* __restrict__ w, int ld,
                                        int n0, int k0) {
  int lane = threadIdx.x & 31;
  const bf16* p = w + (long)(n0 + (lane & 15)) * ld + k0 + ((lane >> 4) << 3);
  v8bf lo = *(const v8bf*)p;
  v8bf hi = *(const v8bf*)(p + 16);
  v16bf r;
#pragma unroll
  for (int i = 0; i < 8; ++i) { r[i] = lo[i]; r[i + 8] = hi[i]; }
  return r;
}

// ---- LDS fragment loaders (tile layout [row][TSTRIDE]) --------------------
__device__ __forceinline__ v16bf frag_a_lds(const bf16* sA, int m0) {
  int lane = threadIdx.x & 31;
  const bf16* p = sA + (m0 + (lane & 15)) * TSTRIDE + ((lane >> 4) << 3);
  v8bf lo = *(const v8bf*)p;
  v8bf hi = *(const v8bf*)(p + 16);
  v16bf r;
#pragma unroll
  for (int i = 0; i < 8; ++i) { r[i] = lo[i]; r[i + 8] = hi[i]; }
  return r;
}

__device__ __forceinline__ v16bf frag_b_lds(const bf16* sB, int n0) {
  int lane = threadIdx.x & 31;
  const bf16* p = sB + (n0 + (lane & 15)) * TSTRIDE + ((lane >> 4) << 3);
  v8bf lo = *(const v8bf*)p;
  v8bf hi = *(const v8bf*)(p + 16);
  v16bf r;
#pragma unroll
  for (int i = 0; i < 8; ++i) { r[i] = lo[i]; r[i + 8] = hi[i]; }
  return r;
}

// ---- async DMA of one 128x32 A tile + 128x32 B tile into LDS --------------
// Each of 256 threads issues 2+2 global_load_async_to_lds_b128 (tracked by
// ASYNCcnt). LDS byte offset = low 32 bits of the flat shared-aperture addr.
__device__ __forceinline__ void async_ldsoff_copy16(unsigned ldsoff,
                                                    const bf16* g) {
  asm volatile("global_load_async_to_lds_b128 %0, %1, off"
               :: "v"(ldsoff), "v"(g)
               : "memory");
}

__device__ __forceinline__ void async_copy_chunk(
    const bf16* __restrict__ A, const bf16* __restrict__ Bw, int K, int mbb,
    int nbb, int k0, bf16* sA, bf16* sB) {
  int t = threadIdx.x;
#pragma unroll
  for (int i = 0; i < 2; ++i) {
    int seg = t + i * 256;            // 512 segments of 16B: row = seg/4
    int row = seg >> 2, s = seg & 3;
    async_ldsoff_copy16(
        (unsigned)(uintptr_t)(sA + row * TSTRIDE + s * 8),
        A + (long)(mbb + row) * K + k0 + s * 8);
  }
#pragma unroll
  for (int i = 0; i < 2; ++i) {
    int seg = t + i * 256;
    int col = seg >> 2, s = seg & 3;
    async_ldsoff_copy16(
        (unsigned)(uintptr_t)(sB + col * TSTRIDE + s * 8),
        Bw + (long)(nbb + col) * K + k0 + s * 8);
  }
}

__device__ __forceinline__ float wave_sum(float v) {
#pragma unroll
  for (int m = 1; m <= 16; m <<= 1) v += __shfl_xor(v, m, 32);
  return v;
}

// ---------------------------------------------------------------------------
__global__ void f32_to_bf16_kernel(const float* __restrict__ in,
                                   bf16* __restrict__ out, int n) {
  int i = blockIdx.x * 256 + threadIdx.x;
  if (i < n) out[i] = (bf16)in[i];
}

// ---------------------------------------------------------------------------
// LN1 fused with roll(-4,-4) + window partition. One wave per output row.
// ---------------------------------------------------------------------------
__global__ __launch_bounds__(256) void ln1_window_kernel(
    const float* __restrict__ x, const float* __restrict__ g,
    const float* __restrict__ b, bf16* __restrict__ xwin) {
  int wid = threadIdx.x >> 5, lane = threadIdx.x & 31;
  long m = (long)blockIdx.x * 8 + wid;
  int win = (int)(m >> 6), l = (int)(m & 63);
  int n = win >> 6, wrem = win & 63;
  int hs = (wrem >> 3) * 8 + (l >> 3);
  int wss = (wrem & 7) * 8 + (l & 7);
  int h = (hs + 4) & 63, w = (wss + 4) & 63;
  const float* row = x + (((long)n * 64 + h) * 64 + w) * 512;

  float vals[16], s = 0.f, s2 = 0.f;
#pragma unroll
  for (int i = 0; i < 16; ++i) {
    float v = row[lane + 32 * i];
    vals[i] = v; s += v; s2 += v * v;
  }
  s = wave_sum(s); s2 = wave_sum(s2);
  float mu = s * (1.f / 512.f);
  float var = s2 * (1.f / 512.f) - mu * mu;
  float rs = rsqrtf(var + 1e-5f);

  bf16* orow = xwin + m * 512;
#pragma unroll
  for (int i = 0; i < 16; ++i) {
    int c = lane + 32 * i;
    orow[c] = (bf16)((vals[i] - mu) * rs * g[c] + b[c]);
  }
}

// ---------------------------------------------------------------------------
// LN2 fused residual: y2 = y1 + LN(y1); writes f32 residual + bf16 MLP input.
// ---------------------------------------------------------------------------
__global__ __launch_bounds__(256) void ln2_kernel(
    const float* __restrict__ y1, const float* __restrict__ g,
    const float* __restrict__ b, float* __restrict__ y2f,
    bf16* __restrict__ y2b) {
  int wid = threadIdx.x >> 5, lane = threadIdx.x & 31;
  long m = (long)blockIdx.x * 8 + wid;
  const float* row = y1 + m * 512;
  float vals[16], s = 0.f, s2 = 0.f;
#pragma unroll
  for (int i = 0; i < 16; ++i) {
    float v = row[lane + 32 * i];
    vals[i] = v; s += v; s2 += v * v;
  }
  s = wave_sum(s); s2 = wave_sum(s2);
  float mu = s * (1.f / 512.f);
  float var = s2 * (1.f / 512.f) - mu * mu;
  float rs = rsqrtf(var + 1e-5f);
#pragma unroll
  for (int i = 0; i < 16; ++i) {
    int c = lane + 32 * i;
    float v = vals[i];
    float y = v + (v - mu) * rs * g[c] + b[c];
    y2f[m * 512 + c] = y;
    y2b[m * 512 + c] = (bf16)y;
  }
}

// ---------------------------------------------------------------------------
// WMMA GEMM: C[M,N] = A[M,K](bf16) @ W[N,K]^T(bf16) + bias, f32 accumulate.
// Block tile 128x128 (8 waves; wave tile 32x64 = 8 WMMAs/chunk). A and B
// tiles DMA'd into LDS (double buffered) with async-to-LDS; one barrier and
// one s_wait_asynccnt per 32-wide K chunk; DMA of chunk c+1 overlaps WMMAs.
// ---------------------------------------------------------------------------
enum { EPI_QKV = 0, EPI_Y1 = 1, EPI_MLP1 = 2, EPI_MLP2 = 3 };

template <int EPI>
__global__ __launch_bounds__(256) void gemm_wmma(
    const bf16* __restrict__ A, const bf16* __restrict__ Bw,
    const float* __restrict__ bias, int M, int N, int K,
    bf16* __restrict__ qb, bf16* __restrict__ kb, bf16* __restrict__ vtb,
    float* __restrict__ outf, bf16* __restrict__ outb,
    const float* __restrict__ resid) {
  __shared__ __align__(16) bf16 sA[2][128 * TSTRIDE];
  __shared__ __align__(16) bf16 sB[2][128 * TSTRIDE];

  int wid = threadIdx.x >> 5;
  int mw = (wid & 3) * 32;   // wave row offset inside block tile
  int nw = (wid >> 2) * 64;  // wave col offset inside block tile
  int mbb = blockIdx.x * 128;
  int nbb = blockIdx.y * 128;

  const v8f vzero = {0.f, 0.f, 0.f, 0.f, 0.f, 0.f, 0.f, 0.f};
  v8f acc[2][4] = {{vzero, vzero, vzero, vzero}, {vzero, vzero, vzero, vzero}};

  async_copy_chunk(A, Bw, K, mbb, nbb, 0, sA[0], sB[0]);
  int nchunks = K >> 5;
  for (int c = 0; c < nchunks; ++c) {
    asm volatile("s_wait_asynccnt 0x0" ::: "memory");
    __syncthreads();  // all waves' DMA of chunk c visible; chunk c-1 consumed
    if (c + 1 < nchunks)
      async_copy_chunk(A, Bw, K, mbb, nbb, (c + 1) << 5, sA[(c + 1) & 1],
                       sB[(c + 1) & 1]);
    const bf16* cA = sA[c & 1];
    const bf16* cB = sB[c & 1];
    v16bf a0 = frag_a_lds(cA, mw);
    v16bf a1 = frag_a_lds(cA, mw + 16);
#pragma unroll
    for (int nt = 0; nt < 4; ++nt) {
      v16bf b = frag_b_lds(cB, nw + nt * 16);
      acc[0][nt] = wmma_bf16(a0, b, acc[0][nt]);
      acc[1][nt] = wmma_bf16(a1, b, acc[1][nt]);
    }
  }

  int lane = threadIdx.x & 31;
  int lh = lane >> 4, n2 = lane & 15;
#pragma unroll
  for (int mt = 0; mt < 2; ++mt)
#pragma unroll
    for (int nt = 0; nt < 4; ++nt)
#pragma unroll
      for (int r = 0; r < 8; ++r) {
        int m = mbb + mw + mt * 16 + r + 8 * lh;
        int n = nbb + nw + nt * 16 + n2;
        float v = acc[mt][nt][r] + bias[n];
        if constexpr (EPI == EPI_QKV) {
          int sel = n >> 9, cc = n & 511;
          int head = cc >> 5, d = cc & 31;
          int win = m >> 6, l = m & 63;
          if (sel == 0)       // q scaled by 1/sqrt(HD)
            qb[(((long)win * 16 + head) * 64 + l) * 32 + d] =
                (bf16)(v * 0.17677669529663687f);
          else if (sel == 1)  // k row-major [l, d]
            kb[(((long)win * 16 + head) * 64 + l) * 32 + d] = (bf16)v;
          else                // v transposed [d, l]
            vtb[(((long)win * 16 + head) * 32 + d) * 64 + l] = (bf16)v;
        } else if constexpr (EPI == EPI_Y1) {
          outf[(long)m * N + n] = v;
        } else if constexpr (EPI == EPI_MLP1) {
          float gx = 0.5f * v * (1.0f + erff(v * 0.70710678118654752f));
          outb[(long)m * N + n] = (bf16)gx;
        } else {  // EPI_MLP2: + y2 residual, reverse windows + roll(+4,+4)
          v += resid[(long)m * 512 + n];
          int win = m >> 6, l = m & 63;
          int nn = win >> 6, wrem = win & 63;
          int hs = (wrem >> 3) * 8 + (l >> 3);
          int wss = (wrem & 7) * 8 + (l & 7);
          int hh = (hs + 4) & 63, ww = (wss + 4) & 63;
          long img = ((long)nn * 64 + hh) * 64 + ww;
          outf[img * 512 + n] = v;
        }
      }
}

// ---------------------------------------------------------------------------
// Attention: one block per (window, head); 4 waves, 16-row strip each.
// ---------------------------------------------------------------------------
__global__ __launch_bounds__(128) void attn_kernel(
    const bf16* __restrict__ qb, const bf16* __restrict__ kb,
    const bf16* __restrict__ vtb, bf16* __restrict__ attn_out) {
  __shared__ __align__(16) bf16 lds[4][16][72];

  int wid = threadIdx.x >> 5, lane = threadIdx.x & 31;
  int win = blockIdx.x >> 4, head = blockIdx.x & 15;
  const bf16* q  = qb  + ((long)win * 16 + head) * 64 * 32;
  const bf16* kk = kb  + ((long)win * 16 + head) * 64 * 32;
  const bf16* vt = vtb + ((long)win * 16 + head) * 32 * 64;
  int m0 = wid * 16;
  int lh = lane >> 4, n2 = lane & 15;
  const v8f vzero = {0.f, 0.f, 0.f, 0.f, 0.f, 0.f, 0.f, 0.f};

  v16bf aq = frag_a(q, 32, m0, 0);
  v8f sc[4];
#pragma unroll
  for (int t = 0; t < 4; ++t)
    sc[t] = wmma_bf16(aq, frag_b(kk, 32, t * 16, 0), vzero);

#pragma unroll
  for (int r = 0; r < 8; ++r) {
    float mx = fmaxf(fmaxf(sc[0][r], sc[1][r]), fmaxf(sc[2][r], sc[3][r]));
#pragma unroll
    for (int msk = 1; msk <= 8; msk <<= 1) mx = fmaxf(mx, __shfl_xor(mx, msk, 32));
    float e[4], sum = 0.f;
#pragma unroll
    for (int t = 0; t < 4; ++t) { e[t] = __expf(sc[t][r] - mx); sum += e[t]; }
#pragma unroll
    for (int msk = 1; msk <= 8; msk <<= 1) sum += __shfl_xor(sum, msk, 32);
    float inv = 1.f / sum;
    int rowl = r + 8 * lh;
#pragma unroll
    for (int t = 0; t < 4; ++t)
      lds[wid][rowl][t * 16 + n2] = (bf16)(e[t] * inv);
  }
  __syncthreads();

  v8f o[2] = {vzero, vzero};
#pragma unroll
  for (int kt = 0; kt < 2; ++kt) {
    v16bf pa;
    {
      int row = lane & 15, kg = lane >> 4;
      const bf16* p = &lds[wid][row][kt * 32 + kg * 8];
      v8bf lo = *(const v8bf*)p;
      v8bf hi = *(const v8bf*)(p + 16);
#pragma unroll
      for (int i = 0; i < 8; ++i) { pa[i] = lo[i]; pa[i + 8] = hi[i]; }
    }
#pragma unroll
    for (int nt = 0; nt < 2; ++nt)
      o[nt] = wmma_bf16(pa, frag_b(vt, 64, nt * 16, kt * 32), o[nt]);
  }

#pragma unroll
  for (int nt = 0; nt < 2; ++nt)
#pragma unroll
    for (int r = 0; r < 8; ++r) {
      long row = (long)win * 64 + m0 + r + 8 * lh;
      attn_out[row * 512 + head * 32 + nt * 16 + n2] = (bf16)o[nt][r];
    }
}

// ---------------------------------------------------------------------------
extern "C" void kernel_launch(void* const* d_in, const int* in_sizes, int n_in,
                              void* d_out, int out_size, void* d_ws,
                              size_t ws_size, hipStream_t stream) {
  (void)in_sizes; (void)n_in; (void)out_size; (void)ws_size;
  const float* x    = (const float*)d_in[0];
  const float* n1g  = (const float*)d_in[1];
  const float* n1b  = (const float*)d_in[2];
  const float* wqkv = (const float*)d_in[3];
  const float* bqkv = (const float*)d_in[4];
  const float* wout = (const float*)d_in[5];
  const float* bout = (const float*)d_in[6];
  const float* n2g  = (const float*)d_in[7];
  const float* n2b  = (const float*)d_in[8];
  const float* w1   = (const float*)d_in[9];
  const float* b1   = (const float*)d_in[10];
  const float* w2   = (const float*)d_in[11];
  const float* b2   = (const float*)d_in[12];
  float* out = (float*)d_out;
  char* ws = (char*)d_ws;

  const size_t MB = 1024ull * 1024ull;
  bf16*  xwin   = (bf16*)(ws);              // 64MB, reused as attn_out
  bf16*  qb     = (bf16*)(ws + 64 * MB);
  bf16*  kb     = (bf16*)(ws + 128 * MB);
  bf16*  vtb    = (bf16*)(ws + 192 * MB);
  float* y1     = (float*)(ws + 64 * MB);   // reuses q/k after attention
  bf16*  y2b    = (bf16*)(ws + 192 * MB);   // reuses vT
  float* y2f    = (float*)(ws + 256 * MB);
  bf16*  hb     = (bf16*)(ws + 384 * MB);
  bf16*  wqkv_b = (bf16*)(ws + 640 * MB);
  bf16*  wout_b = wqkv_b + 1536 * 512;
  bf16*  w1_b   = wout_b + 512 * 512;
  bf16*  w2_b   = w1_b + 2048 * 512;

  auto cvt = [&](const float* s, bf16* d, int n) {
    f32_to_bf16_kernel<<<(n + 255) / 256, 256, 0, stream>>>(s, d, n);
  };
  cvt(wqkv, wqkv_b, 1536 * 512);
  cvt(wout, wout_b, 512 * 512);
  cvt(w1, w1_b, 2048 * 512);
  cvt(w2, w2_b, 512 * 2048);

  ln1_window_kernel<<<8192, 256, 0, stream>>>(x, n1g, n1b, xwin);

  gemm_wmma<EPI_QKV><<<dim3(512, 12), 256, 0, stream>>>(
      xwin, wqkv_b, bqkv, 65536, 1536, 512, qb, kb, vtb, nullptr, nullptr,
      nullptr);

  attn_kernel<<<16384, 128, 0, stream>>>(qb, kb, vtb, xwin /*attn_out*/);

  gemm_wmma<EPI_Y1><<<dim3(512, 4), 256, 0, stream>>>(
      xwin, wout_b, bout, 65536, 512, 512, nullptr, nullptr, nullptr, y1,
      nullptr, nullptr);

  ln2_kernel<<<8192, 256, 0, stream>>>(y1, n2g, n2b, y2f, y2b);

  gemm_wmma<EPI_MLP1><<<dim3(512, 16), 256, 0, stream>>>(
      y2b, w1_b, b1, 65536, 2048, 512, nullptr, nullptr, nullptr, nullptr, hb,
      nullptr);

  gemm_wmma<EPI_MLP2><<<dim3(512, 4), 256, 0, stream>>>(
      hb, w2_b, b2, 65536, 512, 2048, nullptr, nullptr, nullptr, out, nullptr,
      y2f);
}
